// SG_72997264162977
// MI455X (gfx1250) — matrix operand order repaired and checked by reading, not compiled, not run on hardware
//
#include <hip/hip_runtime.h>

// CDNA5 / gfx1250 WMMA types
typedef __attribute__((ext_vector_type(16))) __bf16 v16bf;
typedef __attribute__((ext_vector_type(8)))  float  v8f;

#define DIM        128
#define NEG        5
#define ROW_STRIDE 12   // 2 + NEG + NEG int64 columns per batch row
#define GROUP      16   // batch elements per wave (one 16x16 WMMA tile)
#define WAVES_PER_BLOCK 8

// softplus(clip(x, -10, 10)) ; note -log_sigmoid(y) == softplus(-y)
__device__ __forceinline__ float softplus_clip(float x) {
    x = fminf(fmaxf(x, -10.0f), 10.0f);
    return __logf(1.0f + __expf(x));
}

// convert two float4 chunks into 8 consecutive bf16 slots of a v16bf
template <int OFF>
__device__ __forceinline__ void put8(v16bf& d, const float4& a, const float4& b) {
    d[OFF + 0] = (__bf16)a.x; d[OFF + 1] = (__bf16)a.y;
    d[OFF + 2] = (__bf16)a.z; d[OFF + 3] = (__bf16)a.w;
    d[OFF + 4] = (__bf16)b.x; d[OFF + 5] = (__bf16)b.y;
    d[OFF + 6] = (__bf16)b.z; d[OFF + 7] = (__bf16)b.w;
}

__global__ __launch_bounds__(256) void sgns_wmma_kernel(
    const long long* __restrict__ data,   // (BATCH, 12) int64
    const float*     __restrict__ emb0,   // (VOCAB+1, 128) f32
    const float*     __restrict__ emb1,   // (VOCAB, 128) f32
    float*           __restrict__ partial) // one partial per block
{
    const int lane  = threadIdx.x & 31;
    const int wave  = threadIdx.x >> 5;
    const int group = blockIdx.x * WAVES_PER_BLOCK + wave;
    const long long base = (long long)group * GROUP;

    const int row16 = lane & 15;   // A-row / B-column owned by this lane
    const int hi    = lane >> 4;   // which half-wave (selects K sub-range)

    // ---------------- A = W matrix: 16 rows of emb0, bf16, 4 K-steps ----------
    const long long e_a   = base + row16;
    const long long w_idx = data[e_a * ROW_STRIDE + 1];
    const float* wrow = emb0 + w_idx * DIM;

    v16bf A[4];
    #pragma unroll
    for (int s = 0; s < 4; ++s) {
        const int kbase = 32 * s;
        // ISA A layout (16-bit, 16x32): low lanes K = kbase+{0..7,16..23},
        // high lanes K = kbase+{8..15,24..31}
        const float4* c0 = reinterpret_cast<const float4*>(wrow + kbase + hi * 8);
        const float4* c1 = reinterpret_cast<const float4*>(wrow + kbase + 16 + hi * 8);
        float4 a0 = c0[0], a1 = c0[1];
        float4 b0 = c1[0], b1 = c1[1];
        put8<0>(A[s], a0, a1);
        put8<8>(A[s], b0, b1);
    }

    // ---------------- 6 products: diag(W x B_p), B columns = ctx / neg rows ----
    const long long e_b = base + row16;          // column owned by this lane
    const long long* drow = data + e_b * ROW_STRIDE;

    float ips[1 + NEG];
    #pragma unroll
    for (int p = 0; p < 1 + NEG; ++p) {
        const long long col_idx = (p == 0) ? drow[0] : drow[1 + p]; // ctx or neg[p-1]
        const float* brow = emb1 + col_idx * DIM;

        v8f acc = {};
        #pragma unroll
        for (int s = 0; s < 4; ++s) {
            // B layout matches A: low lanes K = {0..7,16..23}, high lanes K = {8..15,24..31}
            const float4* q = reinterpret_cast<const float4*>(brow + 32 * s + hi * 8);
            const float4* r = reinterpret_cast<const float4*>(brow + 32 * s + 16 + hi * 8);
            float4 q0 = q[0], q1 = q[1], q2 = r[0], q3 = r[1];
            v16bf B;
            put8<0>(B, q0, q1);
            put8<8>(B, q2, q3);
            acc = __builtin_amdgcn_wmma_f32_16x16x32_bf16(
                /*neg_a=*/false, A[s], /*neg_b=*/false, B,
                /*c_mod=*/(short)0, acc, /*reuse_a=*/false, /*reuse_b=*/false);
        }

        // Diagonal extraction: element i at (vgpr=i, lane=i) for i<8,
        // (vgpr=i-8, lane=i+16) for i>=8  =>  lanes 0..7 and 24..31 own a diag.
        float ip = 0.0f;
        #pragma unroll
        for (int v = 0; v < 8; ++v)
            if (lane == v || lane == v + 24) ip = acc[v];
        ips[p] = ip;
    }

    // ---------------- per-lane loss for owned element ----------------
    const bool owner = (lane < 8) || (lane >= 24);
    float lsum = 0.0f;
    if (owner) {
        const long long e_own = base + ((lane < 8) ? lane : (lane - 16));
        const long long* mrow = data + e_own * ROW_STRIDE + 2 + NEG;
        // pos: -log_sigmoid(clip(ip)) = softplus_clip(-ip)
        lsum = softplus_clip(-ips[0]);
        // neg: -log_sigmoid(clip(-ip)) * mask = softplus_clip(ip) * mask
        #pragma unroll
        for (int k = 0; k < NEG; ++k)
            lsum += softplus_clip(ips[1 + k]) * (float)mrow[k];
    }

    // ---------------- wave32 reduction, then block reduction ----------------
    #pragma unroll
    for (int off = 16; off > 0; off >>= 1)
        lsum += __shfl_xor(lsum, off, 32);

    __shared__ float wsum[WAVES_PER_BLOCK];
    if (lane == 0) wsum[wave] = lsum;
    __syncthreads();
    if (threadIdx.x == 0) {
        float s = 0.0f;
        #pragma unroll
        for (int i = 0; i < WAVES_PER_BLOCK; ++i) s += wsum[i];
        partial[blockIdx.x] = s;
    }
}

__global__ __launch_bounds__(256) void sgns_reduce_kernel(
    const float* __restrict__ partial, float* __restrict__ out, int n)
{
    __shared__ float sm[256];
    float s = 0.0f;
    for (int i = threadIdx.x; i < n; i += 256) s += partial[i];
    sm[threadIdx.x] = s;
    __syncthreads();
    #pragma unroll
    for (int stride = 128; stride > 0; stride >>= 1) {
        if (threadIdx.x < stride) sm[threadIdx.x] += sm[threadIdx.x + stride];
        __syncthreads();
    }
    if (threadIdx.x == 0) out[0] = sm[0];
}

extern "C" void kernel_launch(void* const* d_in, const int* in_sizes, int n_in,
                              void* d_out, int out_size, void* d_ws, size_t ws_size,
                              hipStream_t stream) {
    const long long* data = (const long long*)d_in[0];
    const float*     emb0 = (const float*)d_in[1];
    const float*     emb1 = (const float*)d_in[2];

    const int batch  = in_sizes[0] / ROW_STRIDE;                 // 262144
    const int groups = batch / GROUP;                            // 16384
    const int blocks = groups / WAVES_PER_BLOCK;                 // 2048

    float* partial = (float*)d_ws;                               // blocks * 4 bytes

    sgns_wmma_kernel<<<blocks, 256, 0, stream>>>(data, emb0, emb1, partial);
    sgns_reduce_kernel<<<1, 256, 0, stream>>>(partial, (float*)d_out, blocks);
}
